// GATModel_833223655894
// MI455X (gfx1250) — matrix operand order repaired and checked by reading, not compiled, not run on hardware
//
#include <hip/hip_runtime.h>

#define NN 50000
#define EE 800000
#define INCH 128
#define OUTCH 32
#define HEADS 4
#define HC 128            // HEADS*OUTCH
#define NEG_SLOPE 0.2f
#define EPSQ 1e-16f

typedef __attribute__((ext_vector_type(16))) _Float16 v16h;
typedef __attribute__((ext_vector_type(8)))  float    v8f;

// ---- monotonic float<->uint mapping for atomicMax on floats ----
__device__ __forceinline__ unsigned fmap(float x) {
    unsigned b = __float_as_uint(x);
    return (b & 0x80000000u) ? ~b : (b | 0x80000000u);
}
__device__ __forceinline__ float funmap(unsigned m) {
    unsigned b = (m & 0x80000000u) ? (m & 0x7fffffffu) : ~m;
    return __uint_as_float(b);
}
__device__ __forceinline__ float lrelu(float a) { return a > 0.f ? a : NEG_SLOPE * a; }

// ---------------------------------------------------------------------------
// K0: pack W1/W2 into WMMA B-fragment order (f16); fold edge-type path into
//     te1[16][4], te2[16].
// B-fragment k order used here (A matches): lane = g*16+n holds K = g*16 + i (i = 0..15 halves)
// Packed index: (((kk*NT + wt)*32 + lane)*16 + i)
// ---------------------------------------------------------------------------
__global__ void prep_k(const float* __restrict__ W1, const float* __restrict__ W2,
                       const float* __restrict__ emb,
                       const float* __restrict__ We1, const float* __restrict__ ae1,
                       const float* __restrict__ We2, const float* __restrict__ ae2,
                       _Float16* __restrict__ W1p, _Float16* __restrict__ W2p,
                       float* __restrict__ te1, float* __restrict__ te2) {
    int gid = blockIdx.x * 256 + threadIdx.x;
    int stride = gridDim.x * 256;
    // W1: [kk=4][wt=8][lane=32][i=16]
    for (int p = gid; p < INCH * HC; p += stride) {
        int i    = p & 15;
        int lane = (p >> 4) & 31;
        int wt   = (p >> 9) & 7;
        int kk   = p >> 12;
        int k = kk * 32 + (lane >> 4) * 16 + i;
        int n = wt * 16 + (lane & 15);
        W1p[p] = (_Float16)W1[k * HC + n];
    }
    // W2: [kk=4][wt=2][lane=32][i=16]
    for (int p = gid; p < HC * OUTCH; p += stride) {
        int i    = p & 15;
        int lane = (p >> 4) & 31;
        int wt   = (p >> 9) & 1;
        int kk   = p >> 10;
        int k = kk * 32 + (lane >> 4) * 16 + i;
        int n = wt * 16 + (lane & 15);
        W2p[p] = (_Float16)W2[k * OUTCH + n];
    }
    if (gid < 16 * HEADS) {              // te1[t][h] = (emb[t]@We1[:,h,:]) . ae1[h]
        int t = gid >> 2, h = gid & 3;
        float s = 0.f;
        for (int c = 0; c < OUTCH; ++c) {
            float hv = 0.f;
            for (int k = 0; k < OUTCH; ++k) hv += emb[t * OUTCH + k] * We1[k * HC + h * OUTCH + c];
            s += hv * ae1[h * OUTCH + c];
        }
        te1[gid] = s;
    }
    if (gid < 16) {                      // te2[t] = (emb[t]@We2) . ae2
        int t = gid;
        float s = 0.f;
        for (int c = 0; c < OUTCH; ++c) {
            float hv = 0.f;
            for (int k = 0; k < OUTCH; ++k) hv += emb[t * OUTCH + k] * We2[k * OUTCH + c];
            s += hv * ae2[c];
        }
        te2[t] = s;
    }
}

// ---------------------------------------------------------------------------
// K1: H1 = x @ W1   (f32 -> f16 WMMA, f32 accumulate)  [N,128]x[128,128]
// block: 256 threads = 8 waves; each wave owns one 16x16 N-tile; K=128 in 4 steps
// B fragments come pre-packed: one aligned 32B vector load per lane per step.
// ---------------------------------------------------------------------------
__global__ void gemm1_wmma(const float* __restrict__ x, const _Float16* __restrict__ Bp_,
                           float* __restrict__ out) {
    __shared__ _Float16 As[16 * 136];   // padded rows
    const v16h* Bp = (const v16h*)Bp_;
    int tid = threadIdx.x;
    int m0 = blockIdx.x * 16;
    for (int i = tid; i < 16 * 128; i += 256) {
        int r = i >> 7, c = i & 127;
        As[r * 136 + c] = (_Float16)x[(m0 + r) * INCH + c];
    }
    __syncthreads();
    int lane = tid & 31, wave = tid >> 5;
    int cn = wave * 16;
    int nn = lane & 15;     // A row within tile / D column
    int g  = lane >> 4;     // lane half
    v8f acc = {};
    for (int kk = 0; kk < 4; ++kk) {
        int k0 = kk * 32;
        v16h a;
#pragma unroll
        for (int i = 0; i < 16; ++i) {
            int j = i >> 1;
            int ka = k0 + g * 16 + i;
            a[i] = As[nn * 136 + ka];
        }
        v16h b = Bp[(kk * 8 + wave) * 32 + lane];
        acc = __builtin_amdgcn_wmma_f32_16x16x32_f16(false, a, false, b, (short)0, acc, false, false);
    }
#pragma unroll
    for (int r = 0; r < 8; ++r)
        out[(m0 + r + 8 * g) * HC + cn + nn] = acc[r];
}

// ---------------------------------------------------------------------------
// K2: per-node attention dots, layer 1:  s_src[n,h], s_dst[n,h]
// ---------------------------------------------------------------------------
__global__ void dots1_k(const float* __restrict__ H1, const float* __restrict__ as,
                        const float* __restrict__ ad, float* __restrict__ ssrc,
                        float* __restrict__ sdst) {
    int i = blockIdx.x * 256 + threadIdx.x;
    if (i >= NN * HEADS) return;
    int n = i >> 2, h = i & 3;
    const float* hp = H1 + n * HC + h * OUTCH;
    float s1 = 0.f, s2 = 0.f;
    for (int c = 0; c < OUTCH; ++c) { float v = hp[c]; s1 += v * as[h * OUTCH + c]; s2 += v * ad[h * OUTCH + c]; }
    ssrc[i] = s1; sdst[i] = s2;
}

// ---------------------------------------------------------------------------
// K3: edge logits + leaky-relu + segment max (layer 1)
// ---------------------------------------------------------------------------
__global__ void alpha1_k(const int* __restrict__ src, const int* __restrict__ dst,
                         const int* __restrict__ etype, const float* __restrict__ ssrc,
                         const float* __restrict__ sdst, const float* __restrict__ te1,
                         float* __restrict__ alphaE, unsigned* __restrict__ amax) {
    int i = blockIdx.x * 256 + threadIdx.x;
    if (i >= EE * HEADS) return;
    int e = i >> 2, h = i & 3;
    int s = src[e], d = dst[e], t = etype[e];
    float a = lrelu(ssrc[s * HEADS + h] + sdst[d * HEADS + h] + te1[t * HEADS + h]);
    alphaE[i] = a;
    atomicMax(&amax[d * HEADS + h], fmap(a));
}

// ---------------------------------------------------------------------------
// K4: exp(alpha - max) + segment sum (layer 1)
// ---------------------------------------------------------------------------
__global__ void exp1_k(const int* __restrict__ dst, float* __restrict__ alphaE,
                       const unsigned* __restrict__ amax, float* __restrict__ denom) {
    int i = blockIdx.x * 256 + threadIdx.x;
    if (i >= EE * HEADS) return;
    int e = i >> 2, h = i & 3;
    int d = dst[e];
    float ex = expf(alphaE[i] - funmap(amax[d * HEADS + h]));
    alphaE[i] = ex;
    atomicAdd(&denom[d * HEADS + h], ex);
}

// ---------------------------------------------------------------------------
// K5: aggregation layer 1: agg[dst, idx] += H1[src, idx] * w   (2 edges / block)
// ---------------------------------------------------------------------------
__global__ void agg1_k(const int* __restrict__ src, const int* __restrict__ dst,
                       const float* __restrict__ H1, const float* __restrict__ alphaE,
                       const float* __restrict__ denom, float* __restrict__ agg) {
    int tid = threadIdx.x;
    int e = blockIdx.x * 2 + (tid >> 7);
    if (e >= EE) return;
    int idx = tid & 127;
    int h = idx >> 5;
    int s = src[e], d = dst[e];
    float w = alphaE[e * HEADS + h] / (denom[d * HEADS + h] + EPSQ);
    atomicAdd(&agg[d * HC + idx], H1[s * HC + idx] * w);
}

// ---------------------------------------------------------------------------
// K6: relu(agg + b1) -> f16 for second GEMM
// ---------------------------------------------------------------------------
__global__ void relu_bias_k(const float* __restrict__ agg, const float* __restrict__ b1,
                            _Float16* __restrict__ out) {
    int i = blockIdx.x * 256 + threadIdx.x;
    if (i >= NN * HC) return;
    float v = agg[i] + b1[i & 127];
    out[i] = (_Float16)(v > 0.f ? v : 0.f);
}

// ---------------------------------------------------------------------------
// K7: H2 = hrelu @ W2  [N,128]x[128,32]; block: 64 threads = 2 waves
// ---------------------------------------------------------------------------
__global__ void gemm2_wmma(const _Float16* __restrict__ Ah, const _Float16* __restrict__ Bp_,
                           float* __restrict__ out) {
    __shared__ _Float16 As[16 * 136];
    const v16h* Bp = (const v16h*)Bp_;
    int tid = threadIdx.x;
    int m0 = blockIdx.x * 16;
    for (int i = tid; i < 16 * 128; i += 64) {
        int r = i >> 7, c = i & 127;
        As[r * 136 + c] = Ah[(m0 + r) * HC + c];
    }
    __syncthreads();
    int lane = tid & 31, wave = tid >> 5;
    int cn = wave * 16;
    int nn = lane & 15;
    int g  = lane >> 4;
    v8f acc = {};
    for (int kk = 0; kk < 4; ++kk) {
        int k0 = kk * 32;
        v16h a;
#pragma unroll
        for (int i = 0; i < 16; ++i) {
            int j = i >> 1;
            int ka = k0 + g * 16 + i;
            a[i] = As[nn * 136 + ka];
        }
        v16h b = Bp[(kk * 2 + wave) * 32 + lane];
        acc = __builtin_amdgcn_wmma_f32_16x16x32_f16(false, a, false, b, (short)0, acc, false, false);
    }
#pragma unroll
    for (int r = 0; r < 8; ++r)
        out[(m0 + r + 8 * g) * OUTCH + cn + nn] = acc[r];
}

// ---------------------------------------------------------------------------
// K8..K11: layer-2 attention (H=1)
// ---------------------------------------------------------------------------
__global__ void dots2_k(const float* __restrict__ H2, const float* __restrict__ as,
                        const float* __restrict__ ad, float* __restrict__ ssrc,
                        float* __restrict__ sdst) {
    int n = blockIdx.x * 256 + threadIdx.x;
    if (n >= NN) return;
    const float* hp = H2 + n * OUTCH;
    float s1 = 0.f, s2 = 0.f;
    for (int c = 0; c < OUTCH; ++c) { float v = hp[c]; s1 += v * as[c]; s2 += v * ad[c]; }
    ssrc[n] = s1; sdst[n] = s2;
}

__global__ void alpha2_k(const int* __restrict__ src, const int* __restrict__ dst,
                         const int* __restrict__ etype, const float* __restrict__ ssrc,
                         const float* __restrict__ sdst, const float* __restrict__ te2,
                         float* __restrict__ alphaE, unsigned* __restrict__ amax) {
    int e = blockIdx.x * 256 + threadIdx.x;
    if (e >= EE) return;
    int s = src[e], d = dst[e], t = etype[e];
    float a = lrelu(ssrc[s] + sdst[d] + te2[t]);
    alphaE[e] = a;
    atomicMax(&amax[d], fmap(a));
}

__global__ void exp2_k(const int* __restrict__ dst, float* __restrict__ alphaE,
                       const unsigned* __restrict__ amax, float* __restrict__ denom) {
    int e = blockIdx.x * 256 + threadIdx.x;
    if (e >= EE) return;
    int d = dst[e];
    float ex = expf(alphaE[e] - funmap(amax[d]));
    alphaE[e] = ex;
    atomicAdd(&denom[d], ex);
}

__global__ void init_out_k(float* __restrict__ out, const float* __restrict__ b2) {
    int i = blockIdx.x * 256 + threadIdx.x;
    if (i >= NN * OUTCH) return;
    out[i] = b2[i & 31];
}

__global__ void agg2_k(const int* __restrict__ src, const int* __restrict__ dst,
                       const float* __restrict__ H2, const float* __restrict__ alphaE,
                       const float* __restrict__ denom, float* __restrict__ out) {
    int tid = threadIdx.x;
    int e = blockIdx.x * 8 + (tid >> 5);
    if (e >= EE) return;
    int c = tid & 31;
    int s = src[e], d = dst[e];
    float w = alphaE[e] / (denom[d] + EPSQ);
    atomicAdd(&out[d * OUTCH + c], H2[s * OUTCH + c] * w);
}

// ---------------------------------------------------------------------------
extern "C" void kernel_launch(void* const* d_in, const int* in_sizes, int n_in,
                              void* d_out, int out_size, void* d_ws, size_t ws_size,
                              hipStream_t stream) {
    const float* x   = (const float*)d_in[0];
    const int*   ei  = (const int*)  d_in[1];
    const int*   ea  = (const int*)  d_in[2];
    const float* emb = (const float*)d_in[3];
    const float* W1  = (const float*)d_in[4];
    const float* as1 = (const float*)d_in[5];
    const float* ad1 = (const float*)d_in[6];
    const float* We1 = (const float*)d_in[7];
    const float* ae1 = (const float*)d_in[8];
    const float* b1  = (const float*)d_in[9];
    const float* W2  = (const float*)d_in[10];
    const float* as2 = (const float*)d_in[11];
    const float* ad2 = (const float*)d_in[12];
    const float* We2 = (const float*)d_in[13];
    const float* ae2 = (const float*)d_in[14];
    const float* b2  = (const float*)d_in[15];
    const int* src = ei;
    const int* dst = ei + EE;
    float* out = (float*)d_out;

    char* w = (char*)d_ws;
    auto alloc = [&](size_t bytes) -> char* {
        char* p = w; w += (bytes + 255) & ~(size_t)255; return p;
    };
    _Float16* W1p    = (_Float16*)alloc(INCH * HC * 2);
    _Float16* W2p    = (_Float16*)alloc(HC * OUTCH * 2);
    float*    te1    = (float*)   alloc(16 * HEADS * 4);
    float*    te2    = (float*)   alloc(16 * 4);
    float*    H1     = (float*)   alloc((size_t)NN * HC * 4);
    float*    ssrc1  = (float*)   alloc((size_t)NN * HEADS * 4);
    float*    sdst1  = (float*)   alloc((size_t)NN * HEADS * 4);
    unsigned* amax1  = (unsigned*)alloc((size_t)NN * HEADS * 4);
    float*    denom1 = (float*)   alloc((size_t)NN * HEADS * 4);
    float*    alphaE1= (float*)   alloc((size_t)EE * HEADS * 4);
    float*    agg1   = (float*)   alloc((size_t)NN * HC * 4);
    _Float16* hrelu  = (_Float16*)alloc((size_t)NN * HC * 2);
    float*    H2     = (float*)   alloc((size_t)NN * OUTCH * 4);
    float*    ssrc2  = (float*)   alloc((size_t)NN * 4);
    float*    sdst2  = (float*)   alloc((size_t)NN * 4);
    unsigned* amax2  = (unsigned*)alloc((size_t)NN * 4);
    float*    denom2 = (float*)   alloc((size_t)NN * 4);
    float*    alphaE2= (float*)   alloc((size_t)EE * 4);

    // zero the atomic accumulators (mapped-float 0 == "most negative" sentinel for max)
    hipMemsetAsync(amax1,  0, (size_t)NN * HEADS * 4, stream);
    hipMemsetAsync(denom1, 0, (size_t)NN * HEADS * 4, stream);
    hipMemsetAsync(agg1,   0, (size_t)NN * HC * 4,    stream);
    hipMemsetAsync(amax2,  0, (size_t)NN * 4,         stream);
    hipMemsetAsync(denom2, 0, (size_t)NN * 4,         stream);

    prep_k<<<80, 256, 0, stream>>>(W1, W2, emb, We1, ae1, We2, ae2, W1p, W2p, te1, te2);

    gemm1_wmma<<<NN / 16, 256, 0, stream>>>(x, W1p, H1);
    dots1_k<<<(NN * HEADS + 255) / 256, 256, 0, stream>>>(H1, as1, ad1, ssrc1, sdst1);
    alpha1_k<<<(EE * HEADS + 255) / 256, 256, 0, stream>>>(src, dst, ea, ssrc1, sdst1, te1, alphaE1, amax1);
    exp1_k<<<(EE * HEADS + 255) / 256, 256, 0, stream>>>(dst, alphaE1, amax1, denom1);
    agg1_k<<<EE / 2, 256, 0, stream>>>(src, dst, H1, alphaE1, denom1, agg1);
    relu_bias_k<<<(NN * HC + 255) / 256, 256, 0, stream>>>(agg1, b1, hrelu);

    gemm2_wmma<<<NN / 16, 64, 0, stream>>>(hrelu, W2p, H2);
    dots2_k<<<(NN + 255) / 256, 256, 0, stream>>>(H2, as2, ad2, ssrc2, sdst2);
    alpha2_k<<<(EE + 255) / 256, 256, 0, stream>>>(src, dst, ea, ssrc2, sdst2, te2, alphaE2, amax2);
    exp2_k<<<(EE + 255) / 256, 256, 0, stream>>>(dst, alphaE2, amax2, denom2);
    init_out_k<<<(NN * OUTCH + 255) / 256, 256, 0, stream>>>(out, b2);
    agg2_k<<<EE / 8, 256, 0, stream>>>(src, dst, H2, alphaE2, denom2, out);
}
